// GraphAttentionReadOut_17437567222211
// MI455X (gfx1250) — matrix operand order repaired and checked by reading, not compiled, not run on hardware
//
#include <hip/hip_runtime.h>
#include <hip/hip_bf16.h>
#include <math.h>

// GraphAttentionReadOut for MI455X (gfx1250, wave32, WMMA)
//   N_ATOMS ~1e6, FEA=64, HID=32, NH=3, N_GRAPHS=8192, owner sorted.
// Bandwidth-bound: two passes over atom_feas (256MB each) ~= 22us floor @23.3TB/s.
// Kernel 1 computes h^T = W1^T @ X^T so the atom dim lands on lanes:
//   - A fragments (W1^T) are static per wave
//   - B fragments (X^T) load as contiguous 16-float runs per lane
//   - logits contraction needs only one shfl_xor(16) per head

typedef __attribute__((ext_vector_type(16))) __bf16 v16bf;
typedef __attribute__((ext_vector_type(8)))  float  v8f;

#define FEA 64
#define HID 32
#define NH  3

__device__ __forceinline__ float fast_silu(float x) {
  // x * rcp(1+exp(-x)) : v_exp_f32 + v_rcp_f32, no IEEE divide sequence
  return x * __builtin_amdgcn_rcpf(1.0f + __expf(-x));
}

// ---------------------------------------------------------------------------
// Kernel 1: per-atom MLP  logits = silu(X@W1+b1)@W2+b2  via bf16 WMMA,
// computed transposed: D[32,16] = W1^T(32x64) @ X^T(64x16 tile).
// ---------------------------------------------------------------------------
__global__ __launch_bounds__(256)
void mlp_logits_wmma(const float* __restrict__ X,
                     const float* __restrict__ W1,
                     const float* __restrict__ b1,
                     const float* __restrict__ W2,
                     const float* __restrict__ b2,
                     float* __restrict__ logits,
                     int nAtoms, int nTiles)
{
  const int lane = threadIdx.x & 31;
  const int nlo  = lane & 15;   // column (atom) / A-row within tile
  const int hi   = lane >> 4;   // K-half selector
  const int gwave  = (blockIdx.x * blockDim.x + threadIdx.x) >> 5;
  const int nWaves = (gridDim.x * blockDim.x) >> 5;

  // --- A fragments: W1^T (32x64) as 2 M-tiles (hid) x 2 K-tiles (fea).
  // A k order chosen here (contiguous, not ISA table order): both lane halves hold rows M=0..15; lanes 0-15 carry
  // K 0-15 of the K-tile, lanes 16-31 carry K 16-31 (matches B order).
  v16bf afrag[2][2];   // [mt][kt]
#pragma unroll
  for (int mt = 0; mt < 2; ++mt) {
#pragma unroll
    for (int kt = 0; kt < 2; ++kt) {
      const int col = mt * 16 + nlo;          // hid index = A row
#pragma unroll
      for (int e = 0; e < 8; ++e) {
        const int kA = kt * 32 + hi * 16 + e;       // elems 0..7
        const int kB = kt * 32 + hi * 16 + 8 + e;   // elems 8..15
        afrag[mt][kt][e]     = (__bf16)W1[kA * HID + col];
        afrag[mt][kt][8 + e] = (__bf16)W1[kB * HID + col];
      }
    }
  }

  // Per-lane hid slots: VGPR r of D-tile mt holds hid = mt*16 + hi*8 + r.
  float b1v[2][8], w2v[2][8][NH], b2v[NH];
#pragma unroll
  for (int mt = 0; mt < 2; ++mt) {
#pragma unroll
    for (int r = 0; r < 8; ++r) {
      const int hid = mt * 16 + hi * 8 + r;
      b1v[mt][r] = b1[hid];
#pragma unroll
      for (int h = 0; h < NH; ++h) w2v[mt][r][h] = W2[hid * NH + h];
    }
  }
#pragma unroll
  for (int h = 0; h < NH; ++h) b2v[h] = b2[h];

  for (int tile = gwave; tile < nTiles; tile += nWaves) {
    const int base = tile * 16;
    int atom = base + nlo;
    if (atom >= nAtoms) atom = nAtoms - 1;     // clamp (N%16==0 anyway)
    const float4* xr = (const float4*)(X + (size_t)atom * FEA);

    // --- B fragments: X^T (64x16), 2 K-tiles of 32x16.
    // B layout: col N = nlo (atom); lanes 0-15 hold K kt*32+0..15,
    // lanes 16-31 hold K kt*32+16..31  -> one contiguous 16-float run/lane.
    v16bf bfrag[2];
#pragma unroll
    for (int kt = 0; kt < 2; ++kt) {
      const int kbase = (kt * 32 + hi * 16) >> 2;     // float4 index
      const float4 q0 = xr[kbase + 0];
      const float4 q1 = xr[kbase + 1];
      const float4 q2 = xr[kbase + 2];
      const float4 q3 = xr[kbase + 3];
      bfrag[kt][0]  = (__bf16)q0.x; bfrag[kt][1]  = (__bf16)q0.y;
      bfrag[kt][2]  = (__bf16)q0.z; bfrag[kt][3]  = (__bf16)q0.w;
      bfrag[kt][4]  = (__bf16)q1.x; bfrag[kt][5]  = (__bf16)q1.y;
      bfrag[kt][6]  = (__bf16)q1.z; bfrag[kt][7]  = (__bf16)q1.w;
      bfrag[kt][8]  = (__bf16)q2.x; bfrag[kt][9]  = (__bf16)q2.y;
      bfrag[kt][10] = (__bf16)q2.z; bfrag[kt][11] = (__bf16)q2.w;
      bfrag[kt][12] = (__bf16)q3.x; bfrag[kt][13] = (__bf16)q3.y;
      bfrag[kt][14] = (__bf16)q3.z; bfrag[kt][15] = (__bf16)q3.w;
    }

    // h^T tiles: d0 = hid 0..15, d1 = hid 16..31 (atoms on lanes)
    v8f d0 = {}; v8f d1 = {};
    d0 = __builtin_amdgcn_wmma_f32_16x16x32_bf16(false, afrag[0][0], false, bfrag[0], (short)0, d0, false, false);
    d0 = __builtin_amdgcn_wmma_f32_16x16x32_bf16(false, afrag[0][1], false, bfrag[1], (short)0, d0, false, false);
    d1 = __builtin_amdgcn_wmma_f32_16x16x32_bf16(false, afrag[1][0], false, bfrag[0], (short)0, d1, false, false);
    d1 = __builtin_amdgcn_wmma_f32_16x16x32_bf16(false, afrag[1][1], false, bfrag[1], (short)0, d1, false, false);

    // Per-lane partial logits over this lane's 16 hid slots.
    float p[NH] = {0.f, 0.f, 0.f};
#pragma unroll
    for (int r = 0; r < 8; ++r) {
      const float h0 = fast_silu(d0[r] + b1v[0][r]);
      const float h1 = fast_silu(d1[r] + b1v[1][r]);
#pragma unroll
      for (int h = 0; h < NH; ++h)
        p[h] += h0 * w2v[0][r][h] + h1 * w2v[1][r][h];
    }
    // Lane l and l^16 hold complementary hid sets of the same atom.
#pragma unroll
    for (int h = 0; h < NH; ++h)
      p[h] += __shfl_xor(p[h], 16, 32);

    if (hi == 0) {
      const int arow = base + nlo;
      if (arow < nAtoms) {
#pragma unroll
        for (int h = 0; h < NH; ++h)
          logits[(size_t)arow * NH + h] = p[h] + b2v[h];
      }
    }
  }
}

// ---------------------------------------------------------------------------
// Kernel 2: one workgroup per graph. Binary-search segment bounds, block-
// reduce per-head max and exp-sum, then pool: 192 threads each own one (d,h)
// output element; softmax weights for 64-atom chunks staged in LDS.
// ---------------------------------------------------------------------------
__global__ __launch_bounds__(256)
void seg_softmax_pool(const float* __restrict__ X,
                      const int* __restrict__ owner,
                      const float* __restrict__ logits,
                      float* __restrict__ out,
                      int nAtoms)
{
  const int g = blockIdx.x;
  const int t = threadIdx.x;

  __shared__ int   sB[2];
  __shared__ float red[NH * 256];
  __shared__ float sMax[NH];
  __shared__ float sRcp[NH];
  __shared__ float wbuf[64 * NH];

  if (t < 2) {
    const int target = g + t;         // lower_bound(owner, target)
    int lo = 0, hb = nAtoms;
    while (lo < hb) {
      const int mid = (lo + hb) >> 1;
      if (owner[mid] < target) lo = mid + 1; else hb = mid;
    }
    sB[t] = lo;
  }
  __syncthreads();
  const int start = sB[0];
  const int n = sB[1] - start;

  // ---- Phase A: per-head max over the segment ----
  float m[NH];
#pragma unroll
  for (int h = 0; h < NH; ++h) m[h] = -INFINITY;
  for (int i = t; i < n; i += 256) {
    const float* lp = logits + (size_t)(start + i) * NH;
#pragma unroll
    for (int h = 0; h < NH; ++h) m[h] = fmaxf(m[h], lp[h]);
  }
#pragma unroll
  for (int h = 0; h < NH; ++h) red[h * 256 + t] = m[h];
  __syncthreads();
  for (int s = 128; s > 0; s >>= 1) {
    if (t < s) {
#pragma unroll
      for (int h = 0; h < NH; ++h)
        red[h * 256 + t] = fmaxf(red[h * 256 + t], red[h * 256 + t + s]);
    }
    __syncthreads();
  }
  if (t < NH) sMax[t] = red[t * 256];
  __syncthreads();

  // ---- Phase B: denom = sum exp(logit - max); store reciprocal ----
  float sacc[NH] = {0.f, 0.f, 0.f};
  for (int i = t; i < n; i += 256) {
    const float* lp = logits + (size_t)(start + i) * NH;
#pragma unroll
    for (int h = 0; h < NH; ++h) sacc[h] += __expf(lp[h] - sMax[h]);
  }
#pragma unroll
  for (int h = 0; h < NH; ++h) red[h * 256 + t] = sacc[h];
  __syncthreads();
  for (int s = 128; s > 0; s >>= 1) {
    if (t < s) {
#pragma unroll
      for (int h = 0; h < NH; ++h)
        red[h * 256 + t] += red[h * 256 + t + s];
    }
    __syncthreads();
  }
  if (t < NH) sRcp[t] = 1.0f / red[t * 256];
  __syncthreads();

  // ---- Phase C: out[d,h] = sum_a X[a,d] * w[a,h] ----
  const int d = t & 63;
  const int h = t >> 6;               // active producer if h < 3
  float acc = 0.f;
  for (int c0 = 0; c0 < n; c0 += 64) {
    const int cn = min(64, n - c0);
    __syncthreads();
    if (t < cn * NH) {                // stage softmax weights for this chunk
      const int ai = t / NH, hh = t % NH;
      const float lg = logits[(size_t)(start + c0 + ai) * NH + hh];
      wbuf[t] = __expf(lg - sMax[hh]) * sRcp[hh];
    }
    __syncthreads();
    if (h < NH) {
      const float* xp = X + (size_t)(start + c0) * FEA + d;
      for (int i = 0; i < cn; ++i)
        acc += xp[(size_t)i * FEA] * wbuf[i * NH + h];
    }
  }
  if (h < NH) out[(size_t)g * (FEA * NH) + d * NH + h] = acc;
}

// ---------------------------------------------------------------------------
extern "C" void kernel_launch(void* const* d_in, const int* in_sizes, int n_in,
                              void* d_out, int out_size, void* d_ws, size_t ws_size,
                              hipStream_t stream) {
  (void)n_in; (void)ws_size;
  const float* X     = (const float*)d_in[0];
  const int*   owner = (const int*)  d_in[1];
  const float* W1    = (const float*)d_in[2];
  const float* b1    = (const float*)d_in[3];
  const float* W2    = (const float*)d_in[4];
  const float* b2    = (const float*)d_in[5];
  float* out    = (float*)d_out;
  float* logits = (float*)d_ws;                 // nAtoms * 3 floats (~12 MB)

  const int nAtoms  = in_sizes[1];
  const int nGraphs = out_size / (FEA * NH);
  const int nTiles  = (nAtoms + 15) / 16;

  mlp_logits_wmma<<<1280, 256, 0, stream>>>(X, W1, b1, W2, b2, logits, nAtoms, nTiles);
  seg_softmax_pool<<<nGraphs, 256, 0, stream>>>(X, owner, logits, out, nAtoms);
}
